// VanillaRnn_36644660969529
// MI455X (gfx1250) — matrix operand order repaired and checked by reading, not compiled, not run on hardware
//
#include <hip/hip_runtime.h>
#include <math.h>

// ---------------------------------------------------------------------------
// Vanilla RNN  (V=32000, E=512, H=1024, S=256, B=32)  on gfx1250 (CDNA5)
//   phase 1: Wx[h,t]  = sum_e Wih[h,e] * Embed[X[t],e]        (bf16 WMMA)
//   phase 2: Ht       = tanh(Wx[:,s] + Whh @ Ht)   persistent scan, bf16 WMMA
//   phase 3: out[t,v] = sum_h Who[v,h] * Hs[t,h]              (bf16 WMMA)
// ---------------------------------------------------------------------------

typedef __attribute__((ext_vector_type(16))) __bf16 v16bf;
typedef __attribute__((ext_vector_type(8)))  float  v8f;

#define V_ 32000
#define E_ 512
#define H_ 1024
#define S_ 256
#define B_ 32
#define T_ (S_ * B_)          // 8192

#define TILE_K 32
#define LDSS   40             // LDS row stride (bf16 elems): 32 data + 8 pad, 80B rows

// CDNA5 async global->LDS copies (ASYNCcnt-tracked), if the toolchain has them.
// Builtin signature (from clang diagnostic): (int4 AS1* src, int4 AS3* dst,
// imm offset, imm cpol) where int4 = GCC vector_size(16) int.
#if __has_builtin(__builtin_amdgcn_global_load_async_to_lds_b128) && \
    __has_builtin(__builtin_amdgcn_s_wait_asynccnt)
#define USE_ASYNC_LDS 1
typedef int v4i_vs __attribute__((vector_size(16)));
typedef __attribute__((address_space(1))) v4i_vs* gas_b128_t;
typedef __attribute__((address_space(3))) v4i_vs* las_b128_t;
#define GLOBAL_AS(p) ((gas_b128_t)(uintptr_t)(const void*)(p))
#define LDS_AS(p)    ((las_b128_t)(p))
#endif

// ---- WMMA fragment loaders (layouts per CDNA5 ISA 7.12.2) ------------------
// A 16x32 bf16: lanes 0-15 row M=lane, K = {0..7, 16..23}; lanes 16-31 same M,
//               K = {8..15, 24..31}.  rowbase points at K=0 of this lane's row.
__device__ __forceinline__ v16bf load_frag_a(const __bf16* rowbase, int lane) {
  const __bf16* p = rowbase + (((lane >> 4) & 1) << 3);      // +0 or +8
  union { uint4 q[2]; v16bf v; } u;
  u.q[0] = *reinterpret_cast<const uint4*>(p);               // K  +0..7
  u.q[1] = *reinterpret_cast<const uint4*>(p + 16);          // K +16..23
  return u.v;
}
// B 32x16 bf16: lanes 0-15 col N=lane, K = {0..7, 16..23}; lanes 16-31 same N, K = {8..15, 24..31}.
__device__ __forceinline__ v16bf load_frag_b(const __bf16* rowbase, int lane) {
  const __bf16* p = rowbase + (((lane >> 4) & 1) << 3);      // +0 or +8
  union { uint4 q[2]; v16bf v; } u;
  u.q[0] = *reinterpret_cast<const uint4*>(p);
  u.q[1] = *reinterpret_cast<const uint4*>(p + 16);
  return u.v;
}

// branchless saturating tanh: one v_exp_f32 + one v_rcp_f32, no divergence
__device__ __forceinline__ float fast_tanh(float x) {
  float ax = fabsf(x);
  float t  = __expf(-2.0f * ax);          // (0, 1]
  float y  = (1.0f - t) * __builtin_amdgcn_rcpf(1.0f + t);
  return copysignf(y, x);
}

// ---- small prep kernels ----------------------------------------------------
__global__ __launch_bounds__(256) void cvt_f32_bf16_kernel(
    const float* __restrict__ src, __bf16* __restrict__ dst, int n) {
  int i = blockIdx.x * 256 + threadIdx.x;
  if (i < n) dst[i] = (__bf16)src[i];
}

__global__ __launch_bounds__(256) void zero_bf16_kernel(__bf16* dst, int n) {
  int i = blockIdx.x * 256 + threadIdx.x;
  if (i < n) dst[i] = (__bf16)0.0f;
}

__global__ void init_sync_kernel(int* sync) { *sync = 0; }

// G[t, e] = (bf16) Embed[X[t], e]
__global__ __launch_bounds__(256) void gather_embed_kernel(
    const int* __restrict__ X, const float* __restrict__ Embed,
    __bf16* __restrict__ G) {
  int t = blockIdx.x;
  const float* src = Embed + (size_t)X[t] * E_;
  __bf16* dst = G + (size_t)t * E_;
  for (int e = threadIdx.x; e < E_; e += 256) dst[e] = (__bf16)src[e];
}

// ---- tiled WMMA GEMM:  C[M,N] = A[M,K] * B[N,K]^T  (K contiguous) ----------
// 256 threads / 8 waves; 128x128 macro-tile; double-buffered LDS; each wave
// computes 4x2 16x16 tiles (64x32).  bf16 tiles ride the async-to-LDS DMA
// path when available; the f32 Who tile converts through VGPRs.
template <bool B_IS_F32>
__global__ __launch_bounds__(256) void wmma_gemm_kernel(
    const __bf16* __restrict__ A, const void* __restrict__ Bp,
    float* __restrict__ C, int M, int N, int K) {
  __shared__ __bf16 lA[2][128 * LDSS];
  __shared__ __bf16 lB[2][128 * LDSS];

  const int tid   = threadIdx.x;
  const int lane  = tid & 31;
  const int wid   = tid >> 5;
  const int waveM = wid >> 2;             // 0..1  -> 64 rows
  const int waveN = wid & 3;              // 0..3  -> 32 cols
  const int lrow  = lane & 15;
  const int msel  = ((lane >> 4) & 1) << 3;   // C/D row select (0/8)

  const int mBase = blockIdx.y * 128;
  const int nBase = blockIdx.x * 128;

  const int ldRow = tid >> 1;             // 0..127
  const int ldCol = (tid & 1) << 4;       // 0 or 16

  auto load_stage = [&](int k0, int buf) {
    {  // A tile (bf16 source), 128 rows x 32 K
      const __bf16* src = A + (size_t)(mBase + ldRow) * K + (k0 + ldCol);
      __bf16* dst = &lA[buf][ldRow * LDSS + ldCol];
#if defined(USE_ASYNC_LDS)
      __builtin_amdgcn_global_load_async_to_lds_b128(GLOBAL_AS(src),     LDS_AS(dst),     0, 0);
      __builtin_amdgcn_global_load_async_to_lds_b128(GLOBAL_AS(src + 8), LDS_AS(dst + 8), 0, 0);
#else
      uint4 d0 = *reinterpret_cast<const uint4*>(src);
      uint4 d1 = *reinterpret_cast<const uint4*>(src + 8);
      *reinterpret_cast<uint4*>(dst)     = d0;
      *reinterpret_cast<uint4*>(dst + 8) = d1;
      if (k0 + TILE_K < K) __builtin_prefetch(src + TILE_K, 0, 1);
#endif
    }
    if constexpr (B_IS_F32) {   // B tile: f32 -> bf16 through VGPRs (Who)
      const float* src = (const float*)Bp + (size_t)(nBase + ldRow) * K + (k0 + ldCol);
      union { __bf16 h[16]; uint4 q[2]; } u;
#pragma unroll
      for (int j = 0; j < 16; j += 4) {
        float4 f = *reinterpret_cast<const float4*>(src + j);
        u.h[j + 0] = (__bf16)f.x; u.h[j + 1] = (__bf16)f.y;
        u.h[j + 2] = (__bf16)f.z; u.h[j + 3] = (__bf16)f.w;
      }
      *reinterpret_cast<uint4*>(&lB[buf][ldRow * LDSS + ldCol])     = u.q[0];
      *reinterpret_cast<uint4*>(&lB[buf][ldRow * LDSS + ldCol + 8]) = u.q[1];
      if (k0 + TILE_K < K) __builtin_prefetch(src + TILE_K, 0, 1);
    } else {                    // B tile already bf16
      const __bf16* src = (const __bf16*)Bp + (size_t)(nBase + ldRow) * K + (k0 + ldCol);
      __bf16* dst = &lB[buf][ldRow * LDSS + ldCol];
#if defined(USE_ASYNC_LDS)
      __builtin_amdgcn_global_load_async_to_lds_b128(GLOBAL_AS(src),     LDS_AS(dst),     0, 0);
      __builtin_amdgcn_global_load_async_to_lds_b128(GLOBAL_AS(src + 8), LDS_AS(dst + 8), 0, 0);
#else
      uint4 d0 = *reinterpret_cast<const uint4*>(src);
      uint4 d1 = *reinterpret_cast<const uint4*>(src + 8);
      *reinterpret_cast<uint4*>(dst)     = d0;
      *reinterpret_cast<uint4*>(dst + 8) = d1;
      if (k0 + TILE_K < K) __builtin_prefetch(src + TILE_K, 0, 1);
#endif
    }
  };

  v8f acc[4][2] = {};
  const int nstages = K / TILE_K;

  load_stage(0, 0);             // prologue

  for (int st = 0; st < nstages; ++st) {
#if defined(USE_ASYNC_LDS)
    __builtin_amdgcn_s_wait_asynccnt(0);  // my stage-st async copies landed
#endif
    __syncthreads();            // everyone's copies landed; st-1 consumers done
    if (st + 1 < nstages) load_stage((st + 1) * TILE_K, (st + 1) & 1);

    const int buf = st & 1;
    v16bf af[4], bq[2];
#pragma unroll
    for (int mt = 0; mt < 4; ++mt)
      af[mt] = load_frag_a(&lA[buf][(waveM * 64 + mt * 16 + lrow) * LDSS], lane);
#pragma unroll
    for (int nt = 0; nt < 2; ++nt)
      bq[nt] = load_frag_b(&lB[buf][(waveN * 32 + nt * 16 + lrow) * LDSS], lane);

#pragma unroll
    for (int mt = 0; mt < 4; ++mt)
#pragma unroll
      for (int nt = 0; nt < 2; ++nt)
        acc[mt][nt] = __builtin_amdgcn_wmma_f32_16x16x32_bf16(
            false, af[mt], false, bq[nt], (short)0, acc[mt][nt], false, false);
  }

  // epilogue: coalesced f32 stores (lanes sweep N)
#pragma unroll
  for (int mt = 0; mt < 4; ++mt) {
#pragma unroll
    for (int nt = 0; nt < 2; ++nt) {
      const int gm0 = mBase + waveM * 64 + mt * 16 + msel;
      const int gn  = nBase + waveN * 32 + nt * 16 + lrow;
#pragma unroll
      for (int i = 0; i < 8; ++i)
        C[(size_t)(gm0 + i) * N + gn] = acc[mt][nt][i];
    }
  }
}

// ---- persistent recurrence scan: all 256 steps in ONE kernel ---------------
// 16 blocks x 8 waves = 128 waves; one 16x16 tile of [H=1024, B=32] per wave.
// Steps separated by a device-wide monotonic atomic barrier (16 co-resident
// workgroups), replacing 256 kernel launches.
__global__ __launch_bounds__(256) void rnn_scan_kernel(
    const float* __restrict__ Wx,     // [H, T]
    const __bf16* __restrict__ Whhb,  // [H, H]
    __bf16* __restrict__ h0,          // [B, H]  ping (zero-initialized)
    __bf16* __restrict__ h1,          // [B, H]  pong
    __bf16* __restrict__ Hsb,         // [T, H]
    int* __restrict__ sync) {
  const int tid  = threadIdx.x;
  const int lane = tid & 31;
  const int gwid = (blockIdx.x << 3) | (tid >> 5);  // 0..127
  const int m0   = (gwid >> 1) << 4;                // h tile base
  const int n0   = (gwid & 1) << 4;                 // b tile base
  const int lrow = lane & 15;
  const int msel = ((lane >> 4) & 1) << 3;

  __bf16* hp = h0;
  __bf16* hn = h1;

  for (int s = 0; s < S_; ++s) {
    v8f acc;
#pragma unroll
    for (int i = 0; i < 8; ++i)
      acc[i] = Wx[(size_t)(m0 + i + msel) * T_ + s * B_ + n0 + lrow];

    const __bf16* arow = Whhb + (size_t)(m0 + lrow) * H_;
    const __bf16* brow = hp   + (size_t)(n0 + lrow) * H_;
#pragma unroll 4
    for (int k = 0; k < H_; k += TILE_K) {
      v16bf af = load_frag_a(arow + k, lane);
      v16bf bf = load_frag_b(brow + k, lane);
      acc = __builtin_amdgcn_wmma_f32_16x16x32_bf16(
          false, af, false, bf, (short)0, acc, false, false);
    }

#pragma unroll
    for (int i = 0; i < 8; ++i) {
      const int m = m0 + i + msel;
      const int n = n0 + lrow;
      const __bf16 hv = (__bf16)fast_tanh(acc[i]);
      hn[(size_t)n * H_ + m] = hv;
      Hsb[((size_t)(s * B_ + n)) * H_ + m] = hv;
    }

    // ---- device-wide barrier between steps (monotonic counter) ----
    __threadfence();
    __syncthreads();
    if (tid == 0) {
      __hip_atomic_fetch_add(sync, 1, __ATOMIC_ACQ_REL, __HIP_MEMORY_SCOPE_AGENT);
      const int target = (s + 1) * (int)gridDim.x;
      while (__hip_atomic_load(sync, __ATOMIC_ACQUIRE, __HIP_MEMORY_SCOPE_AGENT) < target) {
        __builtin_amdgcn_s_sleep(8);
      }
    }
    __syncthreads();

    __bf16* tmp = hp; hp = hn; hn = tmp;
  }
}

// ---------------------------------------------------------------------------
extern "C" void kernel_launch(void* const* d_in, const int* in_sizes, int n_in,
                              void* d_out, int out_size, void* d_ws, size_t ws_size,
                              hipStream_t stream) {
  (void)in_sizes; (void)n_in; (void)out_size; (void)ws_size;
  const int*   X     = (const int*)  d_in[0];   // [S,B]
  const float* Embed = (const float*)d_in[1];   // [V,E]
  const float* Wih   = (const float*)d_in[2];   // [H,E]
  const float* Whh   = (const float*)d_in[3];   // [H,H]
  const float* Who   = (const float*)d_in[4];   // [V,H]
  float* out = (float*)d_out;                   // [S,B,V] == [T,V]

  char* ws = (char*)d_ws;
  size_t off = 0;
  __bf16* G    = (__bf16*)(ws + off); off += (size_t)T_ * E_ * 2;  //  8 MB
  __bf16* Wihb = (__bf16*)(ws + off); off += (size_t)H_ * E_ * 2;  //  1 MB
  __bf16* Whhb = (__bf16*)(ws + off); off += (size_t)H_ * H_ * 2;  //  2 MB
  float*  Wx   = (float*) (ws + off); off += (size_t)H_ * T_ * 4;  // 32 MB
  __bf16* Hsb  = (__bf16*)(ws + off); off += (size_t)T_ * H_ * 2;  // 16 MB
  __bf16* h0   = (__bf16*)(ws + off); off += (size_t)B_ * H_ * 2;
  __bf16* h1   = (__bf16*)(ws + off); off += (size_t)B_ * H_ * 2;
  int*    sync = (int*)   (ws + off); off += 256;

  // prep: weight conversion, embedding gather, zero state, reset barrier
  cvt_f32_bf16_kernel<<<(H_ * E_ + 255) / 256, 256, 0, stream>>>(Wih, Wihb, H_ * E_);
  cvt_f32_bf16_kernel<<<(H_ * H_ + 255) / 256, 256, 0, stream>>>(Whh, Whhb, H_ * H_);
  gather_embed_kernel<<<T_, 256, 0, stream>>>(X, Embed, G);
  zero_bf16_kernel<<<(B_ * H_ + 255) / 256, 256, 0, stream>>>(h0, B_ * H_);
  init_sync_kernel<<<1, 1, 0, stream>>>(sync);

  // phase 1: Wx[H,T] = Wihb[H,E] * G[T,E]^T
  wmma_gemm_kernel<false><<<dim3(T_ / 128, H_ / 128), 256, 0, stream>>>(
      Wihb, (const void*)G, Wx, H_, T_, E_);

  // phase 2: persistent scan, all 256 steps in one launch
  rnn_scan_kernel<<<16, 256, 0, stream>>>(Wx, Whhb, h0, h1, Hsb, sync);

  // phase 3: out[T,V] = Hsb[T,H] * Who[V,H]^T  (537 GFLOP, dominant)
  wmma_gemm_kernel<true><<<dim3(V_ / 128, T_ / 128), 256, 0, stream>>>(
      Hsb, (const void*)Who, out, T_, V_, H_);
}
